// Encoder_21285857919301
// MI455X (gfx1250) — matrix operand order repaired and checked by reading, not compiled, not run on hardware
//
#include <hip/hip_runtime.h>
#include <hip/hip_bf16.h>
#include <stdint.h>

// ---------------------------------------------------------------------------
// MI455X (gfx1250) transformer encoder layer: bf16 WMMA, TDM staging everywhere.
// ---------------------------------------------------------------------------

typedef __attribute__((ext_vector_type(16))) __bf16 v16bf;
typedef __attribute__((ext_vector_type(8)))  float  v8f;
typedef unsigned int v4u __attribute__((ext_vector_type(4)));
typedef int          v8i __attribute__((ext_vector_type(8)));
typedef int          v4i __attribute__((ext_vector_type(4)));

#define WMMA_BF16(A, B, C) \
  __builtin_amdgcn_wmma_f32_16x16x32_bf16(false, (A), false, (B), (short)0, (C), false, false)

#define ENC_B   2
#define ENC_S   2048
#define ENC_H   16
#define ENC_DM  1024
#define ENC_DK  64
#define ENC_DFF 4096
#define ENC_T   (ENC_B * ENC_S)   // 4096 tokens

// ---------------------------------------------------------------------------
// Fragment loaders (fp32 -> bf16 WMMA fragments)
// ---------------------------------------------------------------------------

// A-matrix 16x32 bf16 fragment: element e maps to
// K = kb + 16*half + e  (one contiguous 16-float run)
static __device__ __forceinline__ v16bf load_a_frag(const float* base, int ld,
                                                    int m, int kb, int half) {
  const float* r = base + (size_t)m * ld + kb;
  const float4 x0 = *(const float4*)(r + 16 * half);
  const float4 x1 = *(const float4*)(r + 16 * half + 4);
  const float4 y0 = *(const float4*)(r + 16 * half + 8);
  const float4 y1 = *(const float4*)(r + 16 * half + 12);
  v16bf a;
  a[0]  = (__bf16)x0.x; a[1]  = (__bf16)x0.y; a[2]  = (__bf16)x0.z; a[3]  = (__bf16)x0.w;
  a[4]  = (__bf16)x1.x; a[5]  = (__bf16)x1.y; a[6]  = (__bf16)x1.z; a[7]  = (__bf16)x1.w;
  a[8]  = (__bf16)y0.x; a[9]  = (__bf16)y0.y; a[10] = (__bf16)y0.z; a[11] = (__bf16)y0.w;
  a[12] = (__bf16)y1.x; a[13] = (__bf16)y1.y; a[14] = (__bf16)y1.z; a[15] = (__bf16)y1.w;
  return a;
}

// B-matrix 32x16 fragment from [N,K] row-major (K contiguous), global or LDS.
// lane column n, element e maps to K = kb + 16*half + e.
static __device__ __forceinline__ v16bf load_bt_frag(const float* base, int ld,
                                                     int n, int kb, int half) {
  const float* r = base + (size_t)n * ld + kb + 16 * half;
  const float4 x0 = *(const float4*)(r + 0);
  const float4 x1 = *(const float4*)(r + 4);
  const float4 x2 = *(const float4*)(r + 8);
  const float4 x3 = *(const float4*)(r + 12);
  v16bf b;
  b[0]  = (__bf16)x0.x; b[1]  = (__bf16)x0.y; b[2]  = (__bf16)x0.z; b[3]  = (__bf16)x0.w;
  b[4]  = (__bf16)x1.x; b[5]  = (__bf16)x1.y; b[6]  = (__bf16)x1.z; b[7]  = (__bf16)x1.w;
  b[8]  = (__bf16)x2.x; b[9]  = (__bf16)x2.y; b[10] = (__bf16)x2.z; b[11] = (__bf16)x2.w;
  b[12] = (__bf16)x3.x; b[13] = (__bf16)x3.y; b[14] = (__bf16)x3.z; b[15] = (__bf16)x3.w;
  return b;
}

// B-matrix 32x16 fragment from [K,N] row-major (N contiguous), strided reads.
// element e maps to K-row = 16*half + e.
static __device__ __forceinline__ v16bf load_b_strided(const float* base, int ld,
                                                       int n, int half) {
  const float* p = base + (size_t)(16 * half) * ld + n;
  v16bf b;
#pragma unroll
  for (int e = 0; e < 16; ++e) b[e] = (__bf16)p[(size_t)e * ld];
  return b;
}

// ---------------------------------------------------------------------------
// Tensor Data Mover: DMA a 2-D fp32 tile [rows x kelts] (row stride =
// row_stride elts) from global memory into LDS at lds_off. D# per ISA §8.3-8.5.
// This toolchain's builtin arity is 6:
//   (uint32x4 g0, int32x8 g1, int32x4 g2, int32x4 g3, int32x8 pad, i32 cpol)
// ---------------------------------------------------------------------------
static __device__ __forceinline__ void tdm_load_2d(const float* gptr, uint32_t lds_off,
                                                   uint32_t kelts, uint32_t rows,
                                                   uint32_t row_stride) {
  const uint64_t ga = (uint64_t)(uintptr_t)gptr;
  v4u g0;
  g0[0] = 1u;                                                // count=1, user descriptor
  g0[1] = lds_off;                                           // lds_addr (bytes)
  g0[2] = (uint32_t)(ga & 0xffffffffu);                      // global_addr[31:0]
  g0[3] = (uint32_t)((ga >> 32) & 0x01ffffffu) | (2u << 30); // addr[56:32] | type=2
  v8i g1;
  g1[0] = (int)(2u << 16);                                   // data_size = 2 -> 4 bytes
  g1[1] = (int)((kelts & 0xffffu) << 16);                    // tensor_dim0[15:0]
  g1[2] = (int)((kelts >> 16) | ((rows & 0xffffu) << 16));   // dim0 hi | tensor_dim1 lo
  g1[3] = (int)((rows >> 16) | (kelts << 16));               // dim1 hi | tile_dim0
  g1[4] = (int)(rows & 0xffffu);                             // tile_dim1 (tile_dim2=0)
  g1[5] = (int)row_stride;                                   // tensor_dim0_stride[31:0]
  g1[6] = 0;
  g1[7] = 0;
  const v4i gz4 = {0, 0, 0, 0};
  const v8i gz8 = {0, 0, 0, 0, 0, 0, 0, 0};
  __builtin_amdgcn_tensor_load_to_lds(g0, g1, gz4, gz4, gz8, 0);
}

static __device__ __forceinline__ uint32_t lds_off_of(const void* p) {
  return (uint32_t)(uintptr_t)p;
}

// ---------------------------------------------------------------------------
// Kernel 1: fused QKV projection with TDM-staged weights.
// Block = (which, head, 32-token group); 8 waves = 2 token tiles x 4 DK tiles.
// Weight slab w[h][kb:kb+32][0:64] (8 KB contiguous) is DMAed into LDS.
// ---------------------------------------------------------------------------
__global__ void __launch_bounds__(256)
qkv_proj_kernel(const float* __restrict__ x,
                const float* __restrict__ wq, const float* __restrict__ wk,
                const float* __restrict__ wv,
                float* __restrict__ q, float* __restrict__ k, float* __restrict__ v) {
  __shared__ float wtile[2][32 * 64];          // double-buffered weight slab

  const int tid  = threadIdx.x;
  const int wave = tid >> 5;
  const int lane = tid & 31;
  const int half = lane >> 4;
  const int l16  = lane & 15;

  const int tg    = blockIdx.x & 127;          // token group (32 tokens)
  const int h     = (blockIdx.x >> 7) & 15;    // head
  const int which = blockIdx.x >> 11;          // 0=q 1=k 2=v
  const int mt    = wave >> 2;                 // 0..1 token sub-tile
  const int nt    = wave & 3;                  // 0..3 DK tile
  const int m0    = tg * 32 + mt * 16;

  const float* w   = (which == 0) ? wq : (which == 1) ? wk : wv;
  float*       out = (which == 0) ? q  : (which == 1) ? k  : v;
  const float* wh  = w + (size_t)h * ENC_DM * ENC_DK;

  const uint32_t lds0 = lds_off_of(&wtile[0][0]);
  const uint32_t lds1 = lds_off_of(&wtile[1][0]);

  const int nk = ENC_DM / 32;                  // 32 k-steps
  if (wave == 0) tdm_load_2d(wh, lds0, 32 * 64, 1, 32 * 64);

  v8f acc = {};
  for (int t = 0; t < nk; ++t) {
    if (wave == 0) __builtin_amdgcn_s_wait_tensorcnt(0);
    __syncthreads();
    if (wave == 0 && t + 1 < nk)
      tdm_load_2d(wh + (size_t)(t + 1) * 32 * ENC_DK,
                  (t & 1) ? lds0 : lds1, 32 * 64, 1, 32 * 64);

    const float* lw = wtile[t & 1];            // [32 k-rows][64 cols]
    v16bf a = load_a_frag(x, ENC_DM, m0 + l16, t * 32, half);
    v16bf b = load_b_strided(lw, ENC_DK, nt * 16 + l16, half);
    acc = WMMA_BF16(a, b, acc);
    __syncthreads();
  }

  const int col = nt * 16 + l16;
#pragma unroll
  for (int r = 0; r < 8; ++r) {
    const int tok = m0 + r + 8 * half;
    const int bb  = tok / ENC_S;
    const int ss  = tok - bb * ENC_S;
    out[(((size_t)bb * ENC_H + h) * ENC_S + ss) * ENC_DK + col] = acc[r];
  }
}

// ---------------------------------------------------------------------------
// Kernel 2: flash attention with TDM-staged K/V slabs.
// Block = 8 query tiles of the same (b,h); 8 waves share K/V in LDS.
// ---------------------------------------------------------------------------
__global__ void __launch_bounds__(256)
flash_attn_kernel(const float* __restrict__ q, const float* __restrict__ k,
                  const float* __restrict__ v, const float* __restrict__ mask,
                  float* __restrict__ concat) {
  __shared__ float  kbuf[2][32 * 64];          // 16 KB
  __shared__ float  vbuf[2][32 * 64];          // 16 KB
  __shared__ __bf16 plds[8][16 * 32];          // per-wave P tile, 8 KB

  const int tid  = threadIdx.x;
  const int wave = tid >> 5;
  const int lane = tid & 31;
  const int half = lane >> 4;
  const int l16  = lane & 15;
  __bf16* pl = plds[wave];

  const int qg = blockIdx.x & 15;              // 16 query groups of 128
  const int h  = (blockIdx.x >> 4) & 15;
  const int bb = blockIdx.x >> 8;

  const size_t bh = (size_t)bb * ENC_H + h;
  const float* qp = q + bh * ENC_S * ENC_DK;
  const float* kp = k + bh * ENC_S * ENC_DK;
  const float* vp = v + bh * ENC_S * ENC_DK;
  const int q0 = qg * 128 + wave * 16;

  const v16bf aq0 = load_a_frag(qp, ENC_DK, q0 + l16, 0, half);
  const v16bf aq1 = load_a_frag(qp, ENC_DK, q0 + l16, 32, half);

  float maskq[8];
#pragma unroll
  for (int r = 0; r < 8; ++r) maskq[r] = mask[bb * ENC_S + q0 + r + 8 * half];

  float mrun[8], lrun[8];
  v8f acc[4];
#pragma unroll
  for (int r = 0; r < 8; ++r) { mrun[r] = -3.0e38f; lrun[r] = 0.f; }
#pragma unroll
  for (int c = 0; c < 4; ++c) acc[c] = (v8f){};

  const float inv_dk = 1.0f / (float)ENC_DK;

  const uint32_t koff0 = lds_off_of(&kbuf[0][0]), koff1 = lds_off_of(&kbuf[1][0]);
  const uint32_t voff0 = lds_off_of(&vbuf[0][0]), voff1 = lds_off_of(&vbuf[1][0]);

  const int njt = ENC_S / 32;                  // 64 key steps
  if (wave == 0) {
    tdm_load_2d(kp, koff0, 32 * 64, 1, 32 * 64);
    tdm_load_2d(vp, voff0, 32 * 64, 1, 32 * 64);
  }

  for (int jt = 0; jt < njt; ++jt) {
    const int j = jt * 32;
    if (wave == 0) __builtin_amdgcn_s_wait_tensorcnt(0);
    __syncthreads();
    if (wave == 0 && jt + 1 < njt) {
      tdm_load_2d(kp + (size_t)(jt + 1) * 32 * ENC_DK, (jt & 1) ? koff0 : koff1,
                  32 * 64, 1, 32 * 64);
      tdm_load_2d(vp + (size_t)(jt + 1) * 32 * ENC_DK, (jt & 1) ? voff0 : voff1,
                  32 * 64, 1, 32 * 64);
    }
    const float* kl = kbuf[jt & 1];            // [32 keys][64 d]
    const float* vl = vbuf[jt & 1];            // [32 keys][64 dv]

    // ---- scores for 2 key sub-tiles, K read from LDS (contiguous) ----
    v8f s0 = {}, s1 = {};
    {
      v16bf b0a = load_bt_frag(kl, ENC_DK, l16, 0, half);
      v16bf b0b = load_bt_frag(kl, ENC_DK, l16, 32, half);
      s0 = WMMA_BF16(aq0, b0a, s0);
      s0 = WMMA_BF16(aq1, b0b, s0);
      v16bf b1a = load_bt_frag(kl, ENC_DK, 16 + l16, 0, half);
      v16bf b1b = load_bt_frag(kl, ENC_DK, 16 + l16, 32, half);
      s1 = WMMA_BF16(aq0, b1a, s1);
      s1 = WMMA_BF16(aq1, b1b, s1);
    }
    const float mk0 = mask[bb * ENC_S + j + l16];
    const float mk1 = mask[bb * ENC_S + j + 16 + l16];

    float f0[8], f1[8], p0[8], p1[8];
#pragma unroll
    for (int r = 0; r < 8; ++r) {
      float a0 = s0[r] * inv_dk * maskq[r] * mk0;
      float a1 = s1[r] * inv_dk * maskq[r] * mk1;
      f0[r] = (a0 == 0.f) ? 1e-10f : a0;
      f1[r] = (a1 == 0.f) ? 1e-10f : a1;
    }

    // ---- online softmax row stats (rows live inside 16-lane halves) ----
#pragma unroll
    for (int r = 0; r < 8; ++r) {
      float t = fmaxf(f0[r], f1[r]);
#pragma unroll
      for (int off = 1; off < 16; off <<= 1) t = fmaxf(t, __shfl_xor(t, off, 32));
      const float mn = fmaxf(mrun[r], t);
      const float sc = __expf(mrun[r] - mn);
      p0[r] = __expf(f0[r] - mn);
      p1[r] = __expf(f1[r] - mn);
      float rs = p0[r] + p1[r];
#pragma unroll
      for (int off = 1; off < 16; off <<= 1) rs += __shfl_xor(rs, off, 32);
      lrun[r] = lrun[r] * sc + rs;
      mrun[r] = mn;
#pragma unroll
      for (int c = 0; c < 4; ++c) acc[c][r] *= sc;
    }

    // ---- C-layout P -> LDS -> A-layout bf16 fragment ----
#pragma unroll
    for (int r = 0; r < 8; ++r) {
      const int row = r + 8 * half;
      pl[row * 32 + l16]      = (__bf16)p0[r];
      pl[row * 32 + 16 + l16] = (__bf16)p1[r];
    }
    __syncthreads();
    v16bf ap;
#pragma unroll
    for (int e = 0; e < 8; ++e) {
      ap[e]     = pl[l16 * 32 + 16 * half + e];
      ap[8 + e] = pl[l16 * 32 + 16 * half + 8 + e];
    }
    __syncthreads();

    // ---- O += P(16x32) x V(32x64), V read strided from LDS ----
#pragma unroll
    for (int ct = 0; ct < 4; ++ct) {
      v16bf bv = load_b_strided(vl, ENC_DK, ct * 16 + l16, half);
      acc[ct] = WMMA_BF16(ap, bv, acc[ct]);
    }
    __syncthreads();                           // block done with slab jt
  }

#pragma unroll
  for (int ct = 0; ct < 4; ++ct) {
#pragma unroll
    for (int r = 0; r < 8; ++r) {
      const float o = acc[ct][r] / lrun[r];
      const int ss = q0 + r + 8 * half;
      concat[((size_t)(bb * ENC_S + ss)) * (ENC_H * ENC_DK) + h * ENC_DK + ct * 16 + l16] = o;
    }
  }
}

// ---------------------------------------------------------------------------
// Kernel 3: TDM-staged GEMM  C[M,N] = A[M,K] * Bt[N,K]^T + bias (+ReLU).
// Block = 8 waves = 128x16 output tile; B slab shared via tensor_load_to_lds.
// ---------------------------------------------------------------------------
__global__ void __launch_bounds__(256)
gemm_bt_tdm_kernel(const float* __restrict__ A, const float* __restrict__ Bt,
                   const float* __restrict__ bias, float* __restrict__ C,
                   int M, int N, int K, int relu) {
  __shared__ float btile[2][16 * 32];

  const int tid  = threadIdx.x;
  const int wave = tid >> 5;
  const int lane = tid & 31;
  const int half = lane >> 4;
  const int l16  = lane & 15;

  const int nblk = N >> 4;
  const int bm = blockIdx.x / nblk;
  const int bn = blockIdx.x - bm * nblk;
  const int m0 = bm * 128 + wave * 16;
  const int n0 = bn * 16;

  const float*   brow = Bt + (size_t)n0 * K;
  const uint32_t lds0 = lds_off_of(&btile[0][0]);
  const uint32_t lds1 = lds_off_of(&btile[1][0]);

  const int nk = K >> 5;
  if (wave == 0) tdm_load_2d(brow, lds0, 32, 16, (uint32_t)K);

  v8f acc = {};
  for (int t = 0; t < nk; ++t) {
    if (wave == 0) __builtin_amdgcn_s_wait_tensorcnt(0);
    __syncthreads();
    if (wave == 0 && t + 1 < nk)
      tdm_load_2d(brow + (t + 1) * 32, (t & 1) ? lds0 : lds1, 32, 16, (uint32_t)K);

    const float* lb = btile[t & 1];
    v16bf b = load_bt_frag(lb, 32, l16, 0, half);
    v16bf a = load_a_frag(A, K, m0 + l16, t * 32, half);
    acc = WMMA_BF16(a, b, acc);
    __syncthreads();
  }

  const int col = n0 + l16;
  const float bz = bias[col];
#pragma unroll
  for (int r = 0; r < 8; ++r) {
    float vv = acc[r] + bz;
    if (relu) vv = fmaxf(vv, 0.f);
    const int row = m0 + r + 8 * half;
    C[(size_t)row * N + col] = vv;
  }
}

// ---------------------------------------------------------------------------
// Kernel 4: out = LayerNorm(a + res) * g + b  over last dim D=1024.
// ---------------------------------------------------------------------------
__global__ void __launch_bounds__(256)
add_ln_kernel(const float* __restrict__ a, const float* __restrict__ res,
              const float* __restrict__ g, const float* __restrict__ be,
              float* __restrict__ out) {
  __shared__ float red[256];
  const int t   = blockIdx.x;
  const int tid = threadIdx.x;
  const int D   = ENC_DM;

  float vals[4];
  float s = 0.f;
#pragma unroll
  for (int i = 0; i < 4; ++i) {
    const int c = tid + i * 256;
    vals[i] = a[(size_t)t * D + c] + res[(size_t)t * D + c];
    s += vals[i];
  }
  red[tid] = s;
  __syncthreads();
  for (int off = 128; off > 0; off >>= 1) {
    if (tid < off) red[tid] += red[tid + off];
    __syncthreads();
  }
  const float mu = red[0] * (1.0f / D);
  __syncthreads();

  float s2 = 0.f;
#pragma unroll
  for (int i = 0; i < 4; ++i) {
    const float d = vals[i] - mu;
    s2 += d * d;
  }
  red[tid] = s2;
  __syncthreads();
  for (int off = 128; off > 0; off >>= 1) {
    if (tid < off) red[tid] += red[tid + off];
    __syncthreads();
  }
  const float var = red[0] * (1.0f / D);
  const float rs  = rsqrtf(var + 1e-6f);
#pragma unroll
  for (int i = 0; i < 4; ++i) {
    const int c = tid + i * 256;
    out[(size_t)t * D + c] = (vals[i] - mu) * rs * g[c] + be[c];
  }
}

// ---------------------------------------------------------------------------
// Host launcher
// ---------------------------------------------------------------------------
extern "C" void kernel_launch(void* const* d_in, const int* in_sizes, int n_in,
                              void* d_out, int out_size, void* d_ws, size_t ws_size,
                              hipStream_t stream) {
  (void)in_sizes; (void)n_in; (void)out_size; (void)ws_size;

  const float* x     = (const float*)d_in[0];
  const float* mask  = (const float*)d_in[1];
  const float* w_q   = (const float*)d_in[2];
  const float* w_k   = (const float*)d_in[3];
  const float* w_v   = (const float*)d_in[4];
  const float* w_o   = (const float*)d_in[5];
  const float* b_o   = (const float*)d_in[6];
  const float* w1    = (const float*)d_in[7];
  const float* b1    = (const float*)d_in[8];
  const float* w2    = (const float*)d_in[9];
  const float* b2    = (const float*)d_in[10];
  const float* ln1_g = (const float*)d_in[11];
  const float* ln1_b = (const float*)d_in[12];
  const float* ln2_g = (const float*)d_in[13];
  const float* ln2_b = (const float*)d_in[14];
  float* outp = (float*)d_out;

  float* ws = (float*)d_ws;
  const size_t SEG = (size_t)ENC_T * 1024;
  float* q      = ws;
  float* k      = q + SEG;
  float* v      = k + SEG;
  float* concat = v + SEG;
  float* proj   = concat + SEG;
  float* x1     = proj + SEG;
  float* h1     = x1 + SEG;
  float* f2     = h1 + (size_t)ENC_T * ENC_DFF;

  const int THREADS = 256;

  // 1) QKV projection: 3 * 16 heads * 128 token-groups = 6144 blocks
  qkv_proj_kernel<<<3 * ENC_H * (ENC_T / 32), THREADS, 0, stream>>>(
      x, w_q, w_k, w_v, q, k, v);

  // 2) Flash attention: B * H * (S/128) = 512 blocks
  flash_attn_kernel<<<ENC_B * ENC_H * (ENC_S / 128), THREADS, 0, stream>>>(
      q, k, v, mask, concat);

  // 3) Output projection + bias
  gemm_bt_tdm_kernel<<<(ENC_T / 128) * (ENC_DM / 16), THREADS, 0, stream>>>(
      concat, w_o, b_o, proj, ENC_T, ENC_DM, ENC_H * ENC_DK, 0);

  // 4) x1 = LN(x + proj)
  add_ln_kernel<<<ENC_T, THREADS, 0, stream>>>(x, proj, ln1_g, ln1_b, x1);

  // 5) FFN1 + ReLU
  gemm_bt_tdm_kernel<<<(ENC_T / 128) * (ENC_DFF / 16), THREADS, 0, stream>>>(
      x1, w1, b1, h1, ENC_T, ENC_DFF, ENC_DM, 1);

  // 6) FFN2
  gemm_bt_tdm_kernel<<<(ENC_T / 128) * (ENC_DM / 16), THREADS, 0, stream>>>(
      h1, w2, b2, f2, ENC_T, ENC_DM, ENC_DFF, 0);

  // 7) out = LN(x1 + f2)
  add_ln_kernel<<<ENC_T, THREADS, 0, stream>>>(x1, f2, ln2_g, ln2_b, outp);
}
